// Attention_4629974745706
// MI455X (gfx1250) — compile-verified, matrix operand order checked by reading, not run on hardware
//
#include <hip/hip_runtime.h>
#include <hip/hip_bf16.h>

typedef __attribute__((ext_vector_type(16))) __bf16   v16bf;
typedef __attribute__((ext_vector_type(8)))  __bf16   v8bf;
typedef __attribute__((ext_vector_type(8)))  float    v8f;
typedef __attribute__((ext_vector_type(4)))  uint32_t u32x4;
typedef __attribute__((ext_vector_type(8)))  uint32_t u32x8;

#define D_MODEL 1024
#define N_HEADS 16
#define D_HEAD  64
#define SEQ     2048
#define BATCH   2
#define MTOT    (BATCH * SEQ)    // 4096 rows
#define NQKV    (3 * D_MODEL)    // 3072

// ---- fragment loaders (CDNA5 wave32 16-bit A/B striping) ------------------
// lane L holds row (row0 + L%16); the 16 halves are k0+(L/16)*8 .. +7 and +16.
__device__ __forceinline__ v16bf load_frag(const __bf16* __restrict__ base,
                                           int ld, int row0, int k0) {
  const int lane = threadIdx.x & 31;
  const int g    = lane >> 4;
  const __bf16* p = base + (row0 + (lane & 15)) * ld + k0 + g * 8;
  v8bf lo = *reinterpret_cast<const v8bf*>(p);
  v8bf hi = *reinterpret_cast<const v8bf*>(p + 16);
  return __builtin_shufflevector(lo, hi, 0,1,2,3,4,5,6,7,8,9,10,11,12,13,14,15);
}

// same striping out of a row-major 64x32 bf16 LDS tile (stride 32 halves)
__device__ __forceinline__ v16bf lds_frag(const __bf16* lp, int row0) {
  const int lane = threadIdx.x & 31;
  const int g    = lane >> 4;
  const __bf16* p = lp + (row0 + (lane & 15)) * 32 + g * 8;
  v8bf lo = *reinterpret_cast<const v8bf*>(p);
  v8bf hi = *reinterpret_cast<const v8bf*>(p + 16);
  return __builtin_shufflevector(lo, hi, 0,1,2,3,4,5,6,7,8,9,10,11,12,13,14,15);
}

__device__ __forceinline__ uint32_t lds_offset(void* p) {
  return (uint32_t)(uintptr_t)(__attribute__((address_space(3))) void*)p;
}

// ---- Tensor Data Mover: stage a 64-row x 32-elem bf16 tile of W into LDS --
__device__ __forceinline__ void tdm_issue_w(const __bf16* W, int nrows,
                                            int n0, int kc, uint32_t ldsoff) {
  const uint64_t ga = (uint64_t)(uintptr_t)(W + n0 * D_MODEL + kc);
  u32x4 g0;
  g0[0] = 1u;                                   // count=1, user mode
  g0[1] = ldsoff;                               // lds_addr
  g0[2] = (uint32_t)ga;                         // global_addr[31:0]
  g0[3] = (uint32_t)(ga >> 32) | 0x80000000u;   // addr[56:32] | type=2
  u32x8 g1;
  g1[0] = 0x00010000u;                          // wg_mask=0, data_size=1 (2B)
  g1[1] = (uint32_t)D_MODEL << 16;              // tensor_dim0 lo16 @ [63:48]
  g1[2] = ((uint32_t)nrows & 0xFFFFu) << 16;    // tensor_dim1 lo16 @ [95:80]
  g1[3] = 32u << 16;                            // tile_dim0=32 @ [127:112]
  g1[4] = 64u;                                  // tile_dim1=64 @ [143:128]
  g1[5] = (uint32_t)D_MODEL;                    // tensor_dim0_stride lo32
  g1[6] = 0u;
  g1[7] = 0u;
  asm volatile("tensor_load_to_lds %0, %1" :: "s"(g0), "s"(g1) : "memory");
}

// ---- shared 128x64 GEMM tile loop: A direct from global, B via TDM+LDS ----
// blockDim = 128 (4 waves); wave wv owns rows m0+32*wv (two 16-row A tiles);
// the shared B strip [n0,n0+64) is double-buffered in LDS, DMA overlapped
// with WMMA via TENSORcnt.
__device__ __forceinline__ void gemm_tile_tdm(const __bf16* __restrict__ A,
                                              const __bf16* __restrict__ W,
                                              int m0, int n0, int nrows_w,
                                              __bf16* bufB /* [2][64*32] */,
                                              v8f acc[2][4]) {
  const int wv    = threadIdx.x >> 5;
  const int mrow0 = m0 + wv * 32;

  if (wv == 0) tdm_issue_w(W, nrows_w, n0, 0, lds_offset(bufB));

  for (int kc = 0; kc < D_MODEL; kc += 32) {
    const int cur = (kc >> 5) & 1;
    if (wv == 0) {
      if (kc + 32 < D_MODEL) {
        tdm_issue_w(W, nrows_w, n0, kc + 32,
                    lds_offset(bufB + (cur ^ 1) * (64 * 32)));
        __builtin_amdgcn_s_wait_tensorcnt(1);   // tile `kc` landed
      } else {
        __builtin_amdgcn_s_wait_tensorcnt(0);
      }
    }
    __syncthreads();                            // LDS tile visible to all waves
    const v16bf a0 = load_frag(A, D_MODEL, mrow0, kc);
    const v16bf a1 = load_frag(A, D_MODEL, mrow0 + 16, kc);
    const __bf16* lB = bufB + cur * (64 * 32);
    v16bf bfr[4];
#pragma unroll
    for (int t = 0; t < 4; ++t) bfr[t] = lds_frag(lB, 16 * t);
#pragma unroll
    for (int t = 0; t < 4; ++t)
      acc[0][t] = __builtin_amdgcn_wmma_f32_16x16x32_bf16(
          false, a0, false, bfr[t], (short)0, acc[0][t], false, false);
#pragma unroll
    for (int t = 0; t < 4; ++t)
      acc[1][t] = __builtin_amdgcn_wmma_f32_16x16x32_bf16(
          false, a1, false, bfr[t], (short)0, acc[1][t], false, false);
    __syncthreads();                            // done reading before overwrite
  }
}

__global__ void cvt_f32_bf16(const float* __restrict__ src,
                             __bf16* __restrict__ dst, int n) {
  int i = blockIdx.x * blockDim.x + threadIdx.x;
  if (i < n) dst[i] = (__bf16)src[i];
}

// ---------------- QKV projection: [4096,1024] @ [3072,1024]^T + bias -------
__global__ void __launch_bounds__(128)
qkv_gemm(const __bf16* __restrict__ A, const __bf16* __restrict__ W,
         const float* __restrict__ bias,
         __bf16* __restrict__ qh, __bf16* __restrict__ kh,
         __bf16* __restrict__ vt) {
  __shared__ __bf16 bufB[2][64 * 32];
  const int nblk = NQKV / 64;                   // 48
  const int m0 = (blockIdx.x / nblk) * 128;
  const int n0 = (blockIdx.x % nblk) * 64;
  const int wv = threadIdx.x >> 5;
  const int lane = threadIdx.x & 31, g = lane >> 4, ncol = lane & 15;

  v8f acc[2][4] = {};
  gemm_tile_tdm(A, W, m0, n0, NQKV, &bufB[0][0], acc);

#pragma unroll
  for (int u = 0; u < 2; ++u) {
    const int mrow = m0 + wv * 32 + 16 * u;
#pragma unroll
    for (int t = 0; t < 4; ++t) {
      const int n   = n0 + 16 * t + ncol;
      const int sec = n >> 10;                  // 0:q 1:k 2:v
      const int hd  = (n & 1023) >> 6;
      const int d   = n & 63;
      const float bn = bias[n];
#pragma unroll
      for (int r = 0; r < 8; ++r) {
        const int m  = mrow + g * 8 + r;
        const int bb = m >> 11;
        const int s  = m & (SEQ - 1);
        const __bf16 val = (__bf16)(acc[u][t][r] + bn);
        if (sec == 0)      qh[((bb * N_HEADS + hd) * SEQ + s) * D_HEAD + d] = val;
        else if (sec == 1) kh[((bb * N_HEADS + hd) * SEQ + s) * D_HEAD + d] = val;
        else               vt[((bb * N_HEADS + hd) * D_HEAD + d) * SEQ + s] = val;
      }
    }
  }
}

// ---------------- Causal flash attention, one wave per 16-row q block ------
__global__ void __launch_bounds__(128)
attn_kernel(const __bf16* __restrict__ qh, const __bf16* __restrict__ kh,
            const __bf16* __restrict__ vt, __bf16* __restrict__ inter) {
  __shared__ __bf16 lP[4][16 * 32];
  const int wv   = threadIdx.x >> 5;
  const int wave = blockIdx.x * 4 + wv;
  const int q0   = (wave & 127) * 16;
  const int bh   = wave >> 7;
  const int lane = threadIdx.x & 31, g = lane >> 4, ncol = lane & 15;

  const __bf16* qp = qh + bh * SEQ * D_HEAD;
  const __bf16* kp = kh + bh * SEQ * D_HEAD;
  const __bf16* vp = vt + bh * D_HEAD * SEQ;

  const v16bf aq0 = load_frag(qp, D_HEAD, q0, 0);
  const v16bf aq1 = load_frag(qp, D_HEAD, q0, 32);

  v8f accO[4] = {};
  float rmax[8], rsum[8];
#pragma unroll
  for (int r = 0; r < 8; ++r) { rmax[r] = -3.0e38f; rsum[r] = 0.f; }

  const int kEnd = q0 + 16;
  for (int kc = 0; kc < kEnd; kc += 32) {
    float p[2][8], chmax[8];
#pragma unroll
    for (int r = 0; r < 8; ++r) chmax[r] = -3.0e38f;

#pragma unroll
    for (int j = 0; j < 2; ++j) {
      const int sk0 = kc + 16 * j;
      const v16bf b0 = load_frag(kp, D_HEAD, sk0, 0);
      const v16bf b1 = load_frag(kp, D_HEAD, sk0, 32);
      v8f s = {};
      s = __builtin_amdgcn_wmma_f32_16x16x32_bf16(false, aq0, false, b0,
                                                  (short)0, s, false, false);
      s = __builtin_amdgcn_wmma_f32_16x16x32_bf16(false, aq1, false, b1,
                                                  (short)0, s, false, false);
#pragma unroll
      for (int r = 0; r < 8; ++r) {
        const int m = q0 + g * 8 + r;
        const int n = sk0 + ncol;
        float v = s[r] * 0.125f;
        if (n > m) v = -3.0e38f;
        p[j][r] = v;
        chmax[r] = fmaxf(chmax[r], v);
      }
    }
#pragma unroll
    for (int r = 0; r < 8; ++r)
#pragma unroll
      for (int off = 1; off < 16; off <<= 1)
        chmax[r] = fmaxf(chmax[r], __shfl_xor(chmax[r], off, 32));

    float alpha[8], csum[8];
#pragma unroll
    for (int r = 0; r < 8; ++r) {
      const float nm = fmaxf(rmax[r], chmax[r]);
      alpha[r] = __expf(rmax[r] - nm);
      rmax[r]  = nm;
      const float p0 = __expf(p[0][r] - nm);
      const float p1 = __expf(p[1][r] - nm);
      p[0][r] = p0; p[1][r] = p1;
      csum[r] = p0 + p1;
    }
#pragma unroll
    for (int r = 0; r < 8; ++r) {
#pragma unroll
      for (int off = 1; off < 16; off <<= 1)
        csum[r] += __shfl_xor(csum[r], off, 32);
      rsum[r] = rsum[r] * alpha[r] + csum[r];
    }
#pragma unroll
    for (int t = 0; t < 4; ++t)
#pragma unroll
      for (int r = 0; r < 8; ++r)
        accO[t][r] *= alpha[r];

    __bf16* lp = lP[wv];
#pragma unroll
    for (int j = 0; j < 2; ++j)
#pragma unroll
      for (int r = 0; r < 8; ++r)
        lp[(g * 8 + r) * 32 + 16 * j + ncol] = (__bf16)p[j][r];
    asm volatile("s_wait_dscnt 0x0" ::: "memory");
    v8bf plo = *reinterpret_cast<const v8bf*>(&lp[(lane & 15) * 32 + g * 8]);
    v8bf phi = *reinterpret_cast<const v8bf*>(&lp[(lane & 15) * 32 + g * 8 + 16]);
    v16bf aP = __builtin_shufflevector(plo, phi,
        0,1,2,3,4,5,6,7,8,9,10,11,12,13,14,15);

    v16bf bv[4];
#pragma unroll
    for (int t = 0; t < 4; ++t) bv[t] = load_frag(vp, SEQ, 16 * t, kc);
#pragma unroll
    for (int t = 0; t < 4; ++t)
      accO[t] = __builtin_amdgcn_wmma_f32_16x16x32_bf16(
          false, aP, false, bv[t], (short)0, accO[t], false, false);
  }

  const int bb = bh >> 4, h = bh & 15;
  float inv[8];
#pragma unroll
  for (int r = 0; r < 8; ++r) inv[r] = 1.0f / rsum[r];
#pragma unroll
  for (int t = 0; t < 4; ++t) {
    const int col = h * D_HEAD + 16 * t + ncol;
#pragma unroll
    for (int r = 0; r < 8; ++r) {
      const int s = q0 + g * 8 + r;
      inter[(bb * SEQ + s) * D_MODEL + col] = (__bf16)(accO[t][r] * inv[r]);
    }
  }
}

// ---------------- Output projection: [4096,1024] @ [1024,1024]^T + b_O ----
__global__ void __launch_bounds__(128)
out_gemm(const __bf16* __restrict__ A, const __bf16* __restrict__ W,
         const float* __restrict__ bias, float* __restrict__ out) {
  __shared__ __bf16 bufB[2][64 * 32];
  const int nblk = D_MODEL / 64;                // 16
  const int m0 = (blockIdx.x / nblk) * 128;
  const int n0 = (blockIdx.x % nblk) * 64;
  const int wv = threadIdx.x >> 5;
  const int lane = threadIdx.x & 31, g = lane >> 4, ncol = lane & 15;

  v8f acc[2][4] = {};
  gemm_tile_tdm(A, W, m0, n0, D_MODEL, &bufB[0][0], acc);

#pragma unroll
  for (int u = 0; u < 2; ++u) {
    const int mrow = m0 + wv * 32 + 16 * u;
#pragma unroll
    for (int t = 0; t < 4; ++t) {
      const int n = n0 + 16 * t + ncol;
      const float bn = bias[n];
#pragma unroll
      for (int r = 0; r < 8; ++r) {
        const int m = mrow + g * 8 + r;
        out[m * D_MODEL + n] = acc[u][t][r] + bn;
      }
    }
  }
}

extern "C" void kernel_launch(void* const* d_in, const int* in_sizes, int n_in,
                              void* d_out, int out_size, void* d_ws, size_t ws_size,
                              hipStream_t stream) {
  const float* res    = (const float*)d_in[0];
  const float* W_attn = (const float*)d_in[1];
  const float* b_attn = (const float*)d_in[2];
  const float* W_O    = (const float*)d_in[3];
  const float* b_O    = (const float*)d_in[4];
  float* out = (float*)d_out;

  char* ws = (char*)d_ws;
  __bf16* res_bf   = (__bf16*)(ws);             //  8 MiB: 4096x1024
  __bf16* Wattn_bf = (__bf16*)(ws + 8388608);   //  6 MiB: 3072x1024
  __bf16* WO_bf    = (__bf16*)(ws + 14680064);  //  2 MiB: 1024x1024
  __bf16* q_bf     = (__bf16*)(ws + 16777216);  //  8 MiB: [2,16,2048,64]
  __bf16* k_bf     = (__bf16*)(ws + 25165824);  //  8 MiB
  __bf16* vt_bf    = (__bf16*)(ws + 33554432);  //  8 MiB (transposed V)
  __bf16* inter_bf = (__bf16*)(ws + 41943040);  //  8 MiB

  int n = MTOT * D_MODEL;
  cvt_f32_bf16<<<(n + 255) / 256, 256, 0, stream>>>(res, res_bf, n);
  n = NQKV * D_MODEL;
  cvt_f32_bf16<<<(n + 255) / 256, 256, 0, stream>>>(W_attn, Wattn_bf, n);
  n = D_MODEL * D_MODEL;
  cvt_f32_bf16<<<(n + 255) / 256, 256, 0, stream>>>(W_O, WO_bf, n);

  qkv_gemm<<<(MTOT / 128) * (NQKV / 64), 128, 0, stream>>>(
      res_bf, Wattn_bf, b_attn, q_bf, k_bf, vt_bf);

  attn_kernel<<<(BATCH * N_HEADS * (SEQ / 16)) / 4, 128, 0, stream>>>(
      q_bf, k_bf, vt_bf, inter_bf);

  out_gemm<<<(MTOT / 128) * (D_MODEL / 64), 128, 0, stream>>>(
      inter_bf, WO_bf, b_O, out);
}
